// TAttention_52621939310607
// MI455X (gfx1250) — compile-verified, matrix operand order checked by reading, not run on hardware
//
#include <hip/hip_runtime.h>
#include <hip/hip_bf16.h>
#include <math.h>

// ---------------------------------------------------------------------------
// Transformer attention block on gfx1250 (MI455X), wave32 + WMMA f16.
//   B=4, S=2048, D=1024, H=16, Dh=64.  All matmuls via v_wmma_f32_16x16x32_f16.
// Pipeline: cvt(x) -> transpose(W*) -> GEMM Q,K (row-major f16) + GEMM V
// (epilogue writes V^T per head) -> flash attention with TRANSPOSED score /
// output tiles, 32 queries per wave (K/V fragments reused across two query
// subtiles -> 1 load per WMMA) -> O-proj GEMM (f32) -> residual+LayerNorm.
// ---------------------------------------------------------------------------

typedef __attribute__((ext_vector_type(8)))  _Float16 v8h;
typedef __attribute__((ext_vector_type(16))) _Float16 v16h;
typedef __attribute__((ext_vector_type(8)))  float    v8f;

#define BATCH   4
#define SEQ     2048
#define DMODEL  1024
#define HEADS   16
#define HDIM    64
#define ROWS    (BATCH * SEQ)        // 8192

__device__ __forceinline__ v8f wmma_f16(v16h a, v16h b, v8f c) {
  // D = A(16x32 f16) * B(32x16 f16) + C(16x16 f32)
  return __builtin_amdgcn_wmma_f32_16x16x32_f16(
      /*neg_a=*/false, a, /*neg_b=*/false, b,
      /*c_mod=*/(short)0, c, /*reuse_a=*/false, /*reuse_b=*/false);
}

// Build a 16-halfword fragment from two contiguous 8-halfword chunks
// (ISA layout: per lane, K = [8*h2 .. 8*h2+7] then [16+8*h2 .. 16+8*h2+7]).
__device__ __forceinline__ v16h make_frag(const _Float16* lo, const _Float16* hi) {
  v8h a = *reinterpret_cast<const v8h*>(lo);
  v8h b = *reinterpret_cast<const v8h*>(hi);
  return __builtin_shufflevector(a, b, 0, 1, 2, 3, 4, 5, 6, 7,
                                        8, 9, 10, 11, 12, 13, 14, 15);
}

// ---------------------------------------------------------------------------
__global__ void cvt_f32_f16_kernel(const float* __restrict__ src,
                                   _Float16* __restrict__ dst, int n) {
  int i = blockIdx.x * blockDim.x + threadIdx.x;
  if (i < n) dst[i] = (_Float16)src[i];
}

// W[K][N] f32 -> Wt[N][K] f16  (so GEMM B-fragments read contiguous K)
__global__ void wtrans_kernel(const float* __restrict__ W,
                              _Float16* __restrict__ Wt) {
  int idx = blockIdx.x * blockDim.x + threadIdx.x;     // 1M threads
  int n = idx & (DMODEL - 1);
  int k = idx >> 10;
  Wt[(size_t)n * DMODEL + k] = (_Float16)W[idx];
}

// ---------------------------------------------------------------------------
// WMMA GEMM: out[M x 1024] = A(f16, M x 1024) @ Wt^T + bias, scaled.
// Block tile 64x128, 4 waves (2x2), wave tile 32x64 = 8 v_wmma per K-step
// against 12 LDS fragment loads.
// MODE 0: f16 row-major out.  MODE 1: f32 row-major out.
// MODE 2: f16 out scattered to vt[b][h][d][s] (per-head transposed V).
// ---------------------------------------------------------------------------
template <int MODE>
__global__ __launch_bounds__(128) void gemm_kernel(
    const _Float16* __restrict__ A,   // [M][1024] row-major f16
    const _Float16* __restrict__ Bt,  // [1024][1024] = W^T, row n holds col n of W
    const float* __restrict__ bias,   // [1024]
    _Float16* __restrict__ outH, float* __restrict__ outF, float scale) {
  __shared__ _Float16 As[64][40];     // stride 40 keeps 16B align + spreads banks
  __shared__ _Float16 Bs[128][40];

  const int tid  = threadIdx.x;
  const int lane = tid & 31;
  const int wid  = tid >> 5;
  const int li   = lane & 15;
  const int h2   = lane >> 4;
  const int mw   = (wid >> 1) * 32;   // wave M offset (0/32)
  const int nw   = (wid & 1) * 64;    // wave N offset (0/64)
  const int bm   = blockIdx.x * 64;
  const int bn   = blockIdx.y * 128;

  const int lr = tid >> 1;            // 0..63 : A tile row to stage
  const int lc = (tid & 1) * 16;      // 0/16  : half-row (16 f16 = 32B)

  v8f acc[2][4] = {};

  for (int kb = 0; kb < DMODEL; kb += 32) {
    __syncthreads();
    const v8h* ga = reinterpret_cast<const v8h*>(A + (size_t)(bm + lr) * DMODEL + kb + lc);
    *reinterpret_cast<v8h*>(&As[lr][lc])     = ga[0];
    *reinterpret_cast<v8h*>(&As[lr][lc + 8]) = ga[1];
    // B tile: 128 rows, each thread stages one full row (32 f16 = 64B).
    const v8h* gb = reinterpret_cast<const v8h*>(Bt + (size_t)(bn + tid) * DMODEL + kb);
    *reinterpret_cast<v8h*>(&Bs[tid][0])  = gb[0];
    *reinterpret_cast<v8h*>(&Bs[tid][8])  = gb[1];
    *reinterpret_cast<v8h*>(&Bs[tid][16]) = gb[2];
    *reinterpret_cast<v8h*>(&Bs[tid][24]) = gb[3];
    __syncthreads();

    v16h af[2], bf[4];
#pragma unroll
    for (int mt = 0; mt < 2; ++mt) {
      const _Float16* r = &As[mw + mt * 16 + li][0];
      af[mt] = make_frag(r + 8 * h2, r + 16 + 8 * h2);
    }
#pragma unroll
    for (int nt = 0; nt < 4; ++nt) {
      const _Float16* r = &Bs[nw + nt * 16 + li][0];
      bf[nt] = make_frag(r + 8 * h2, r + 16 + 8 * h2);
    }
#pragma unroll
    for (int mt = 0; mt < 2; ++mt)
#pragma unroll
      for (int nt = 0; nt < 4; ++nt)
        acc[mt][nt] = wmma_f16(af[mt], bf[nt], acc[mt][nt]);
  }

  // Epilogue: D-layout = per-lane column (bn+nw+nt*16+li), rows r + 8*h2.
#pragma unroll
  for (int nt = 0; nt < 4; ++nt) {
    const int col = bn + nw + nt * 16 + li;
    const float bs = bias[col];
#pragma unroll
    for (int mt = 0; mt < 2; ++mt) {
      const int rbase = bm + mw + mt * 16 + 8 * h2;
#pragma unroll
      for (int r = 0; r < 8; ++r) {
        const int row = rbase + r;                     // b*SEQ + s
        const float v = (acc[mt][nt][r] + bs) * scale;
        if (MODE == 0) {
          outH[(size_t)row * DMODEL + col] = (_Float16)v;
        } else if (MODE == 1) {
          outF[(size_t)row * DMODEL + col] = v;
        } else {
          const int bb = row >> 11, s = row & (SEQ - 1);
          const int hh = col >> 6,  d = col & (HDIM - 1);
          outH[((size_t)((bb * HEADS + hh) * HDIM + d) << 11) + s] = (_Float16)v;
        }
      }
    }
  }
}

// ---------------------------------------------------------------------------
// Flash attention with transposed tiles, 32 queries per wave (two 16-query
// subtiles sharing every K/V fragment -> 16 WMMA per 16 b128 loads / key tile).
//   S^T = K_tile x Q^T : lane = query column -> softmax stats lane-local
//                        (one shfl_xor(16) to merge the two key halves).
//   P^T per-lane D-layout == fragment K-chunk layout -> register-only reshape.
//   O^T = V^T x P^T     : lane = query column -> flash rescale is a plain
//                        per-lane multiply; final store is one b128/lane/tile.
// No LDS.
// ---------------------------------------------------------------------------
__global__ __launch_bounds__(32) void attn_kernel(
    const _Float16* __restrict__ qh,  // [b][s][h*64+d], pre-scaled by 1/32
    const _Float16* __restrict__ kh,  // [b][s][h*64+d]
    const _Float16* __restrict__ vt,  // [b][h][d][s]
    _Float16* __restrict__ ctx) {     // [b][s][h*64+d]
  const int lane = threadIdx.x & 31;
  const int li = lane & 15;
  const int h2 = lane >> 4;
  const int bh = blockIdx.y;          // b*16 + h
  const int b = bh >> 4, h = bh & 15;
  const int q0 = blockIdx.x * 32;

  // Q as B-operand: column = query, contiguous d-chunks. Two query subtiles.
  const _Float16* qrow0 = qh + ((size_t)(b * SEQ + q0 + li) * DMODEL) + h * HDIM;
  const _Float16* qrow1 = qrow0 + (size_t)16 * DMODEL;
  const v16h qa0 = make_frag(qrow0 + 8 * h2,      qrow0 + 16 + 8 * h2);
  const v16h qb0 = make_frag(qrow0 + 32 + 8 * h2, qrow0 + 48 + 8 * h2);
  const v16h qa1 = make_frag(qrow1 + 8 * h2,      qrow1 + 16 + 8 * h2);
  const v16h qb1 = make_frag(qrow1 + 32 + 8 * h2, qrow1 + 48 + 8 * h2);

  float m_run[2] = {-INFINITY, -INFINITY};   // per-lane = per-query stats
  float l_run[2] = {0.0f, 0.0f};
  v8f o[2][4] = {};                   // O^T: [q subtile][d tile nt]

  const _Float16* vbase0 = vt + ((size_t)bh * HDIM << 11);

  for (int kt = 0; kt < SEQ / 32; ++kt) {
    const int kbase = kt * 32;

    // K as A-operand: row = key, contiguous d-chunks. Shared by both q tiles.
    const _Float16* kr0 = kh + ((size_t)(b * SEQ + kbase + li) * DMODEL) + h * HDIM;
    const _Float16* kr1 = kr0 + (size_t)16 * DMODEL;
    v16h k0a = make_frag(kr0 + 8 * h2,      kr0 + 16 + 8 * h2);
    v16h k0b = make_frag(kr0 + 32 + 8 * h2, kr0 + 48 + 8 * h2);
    v16h k1a = make_frag(kr1 + 8 * h2,      kr1 + 16 + 8 * h2);
    v16h k1b = make_frag(kr1 + 32 + 8 * h2, kr1 + 48 + 8 * h2);

    // Sequential per-q-subtile softmax keeps only one score block live.
    v16h pf[2];
#pragma unroll
    for (int qq = 0; qq < 2; ++qq) {
      const v16h qa = qq ? qa1 : qa0;
      const v16h qb = qq ? qb1 : qb0;

      // S^T subtiles: st0 = keys kbase+0..15, st1 = +16..31.
      // Lane view: st0[r] <-> key kbase + 8*h2 + r ; st1[r] <-> +16.
      v8f st0 = {}, st1 = {};
      st0 = wmma_f16(k0a, qa, st0);
      st0 = wmma_f16(k0b, qb, st0);
      st1 = wmma_f16(k1a, qa, st1);
      st1 = wmma_f16(k1b, qb, st1);

      // Lane-local max over 16 scores + one half-merge shuffle.
      float mx = fmaxf(st0[0], st1[0]);
#pragma unroll
      for (int r = 1; r < 8; ++r) mx = fmaxf(mx, fmaxf(st0[r], st1[r]));
      mx = fmaxf(mx, __shfl_xor(mx, 16, 32));

      const float m_new = fmaxf(m_run[qq], mx);
      const float corr  = __expf(m_run[qq] - m_new);
      float p0[8], p1[8];
      float sum = 0.0f;
#pragma unroll
      for (int r = 0; r < 8; ++r) {
        p0[r] = __expf(st0[r] - m_new);
        p1[r] = __expf(st1[r] - m_new);
        sum += p0[r] + p1[r];
      }
      sum += __shfl_xor(sum, 16, 32);
      l_run[qq] = l_run[qq] * corr + sum;
      m_run[qq] = m_new;

      // Flash rescale of O^T: corr is per-query = per-lane.
#pragma unroll
      for (int nt = 0; nt < 4; ++nt)
#pragma unroll
        for (int r = 0; r < 8; ++r) o[qq][nt][r] *= corr;

      // P^T fragment: D-layout keys {8h2..+7} U {16+8h2..+7} == B-operand
      // K-chunk layout. Pure register f32->f16 convert.
#pragma unroll
      for (int r = 0; r < 8; ++r) {
        pf[qq][r]     = (_Float16)p0[r];
        pf[qq][r + 8] = (_Float16)p1[r];
      }
    }

    // O^T += V^T x P^T; each V fragment feeds both query subtiles.
#pragma unroll
    for (int nt = 0; nt < 4; ++nt) {
      const _Float16* vrow = vbase0 + ((size_t)(nt * 16 + li) << 11) + kbase;
      v16h vf = make_frag(vrow + 8 * h2, vrow + 16 + 8 * h2);
      o[0][nt] = wmma_f16(vf, pf[0], o[0][nt]);
      o[1][nt] = wmma_f16(vf, pf[1], o[1][nt]);
    }
  }

  // Normalize and store: lane holds query q0 + qq*16 + li, d = nt*16+8*h2+r
  // -> 8 contiguous f16 per tile = one 16B store per tile per lane.
#pragma unroll
  for (int qq = 0; qq < 2; ++qq) {
    const float inv = 1.0f / l_run[qq];
    _Float16* crow = ctx + (size_t)(b * SEQ + q0 + qq * 16 + li) * DMODEL
                   + h * HDIM + 8 * h2;
#pragma unroll
    for (int nt = 0; nt < 4; ++nt) {
      v8h ov;
#pragma unroll
      for (int r = 0; r < 8; ++r) ov[r] = (_Float16)(o[qq][nt][r] * inv);
      *reinterpret_cast<v8h*>(crow + nt * 16) = ov;
    }
  }
}

// ---------------------------------------------------------------------------
// Fused residual + LayerNorm(eps=1e-12): one block per row of 1024.
// ---------------------------------------------------------------------------
__global__ __launch_bounds__(256) void ln_kernel(
    const float* __restrict__ attn, const float* __restrict__ x,
    const float* __restrict__ gamma, const float* __restrict__ beta,
    float* __restrict__ out) {
  __shared__ float red[256];
  const int tid = threadIdx.x;
  const size_t base = (size_t)blockIdx.x * DMODEL;

  float hv[4];
  float s1 = 0.0f;
#pragma unroll
  for (int j = 0; j < 4; ++j) {
    const int c = tid + j * 256;
    hv[j] = attn[base + c] + x[base + c];
    s1 += hv[j];
  }
  red[tid] = s1; __syncthreads();
  for (int st = 128; st > 0; st >>= 1) {
    if (tid < st) red[tid] += red[tid + st];
    __syncthreads();
  }
  const float mu = red[0] * (1.0f / DMODEL);
  __syncthreads();

  float s2 = 0.0f;
#pragma unroll
  for (int j = 0; j < 4; ++j) { const float d = hv[j] - mu; s2 += d * d; }
  red[tid] = s2; __syncthreads();
  for (int st = 128; st > 0; st >>= 1) {
    if (tid < st) red[tid] += red[tid + st];
    __syncthreads();
  }
  const float rstd = rsqrtf(red[0] * (1.0f / DMODEL) + 1e-12f);

#pragma unroll
  for (int j = 0; j < 4; ++j) {
    const int c = tid + j * 256;
    out[base + c] = (hv[j] - mu) * rstd * gamma[c] + beta[c];
  }
}

// ---------------------------------------------------------------------------
extern "C" void kernel_launch(void* const* d_in, const int* in_sizes, int n_in,
                              void* d_out, int out_size, void* d_ws, size_t ws_size,
                              hipStream_t stream) {
  const float* x     = (const float*)d_in[0];
  const float* Wq    = (const float*)d_in[1];
  const float* bq    = (const float*)d_in[2];
  const float* Wk    = (const float*)d_in[3];
  const float* bk    = (const float*)d_in[4];
  const float* Wv    = (const float*)d_in[5];
  const float* bv    = (const float*)d_in[6];
  const float* Wo    = (const float*)d_in[7];
  const float* bo    = (const float*)d_in[8];
  const float* gamma = (const float*)d_in[9];
  const float* beta  = (const float*)d_in[10];
  float* out = (float*)d_out;

  char* ws = (char*)d_ws;
  const size_t MB = (size_t)1 << 20;
  _Float16* xh   = (_Float16*)(ws);                 // 16 MB
  _Float16* wqt  = (_Float16*)(ws + 16  * MB);      //  2 MB each
  _Float16* wkt  = (_Float16*)(ws + 18  * MB);
  _Float16* wvt  = (_Float16*)(ws + 20  * MB);
  _Float16* wot  = (_Float16*)(ws + 22  * MB);
  _Float16* qh   = (_Float16*)(ws + 24  * MB);      // 16 MB each
  _Float16* kh   = (_Float16*)(ws + 40  * MB);
  _Float16* vtp  = (_Float16*)(ws + 56  * MB);      // V^T per head
  _Float16* ctx  = (_Float16*)(ws + 72  * MB);
  float*    attn = (float*)   (ws + 88  * MB);      // 32 MB

  const int NX = BATCH * SEQ * DMODEL;              // 8,388,608

  cvt_f32_f16_kernel<<<NX / 256, 256, 0, stream>>>(x, xh, NX);
  wtrans_kernel<<<(DMODEL * DMODEL) / 256, 256, 0, stream>>>(Wq, wqt);
  wtrans_kernel<<<(DMODEL * DMODEL) / 256, 256, 0, stream>>>(Wk, wkt);
  wtrans_kernel<<<(DMODEL * DMODEL) / 256, 256, 0, stream>>>(Wv, wvt);
  wtrans_kernel<<<(DMODEL * DMODEL) / 256, 256, 0, stream>>>(Wo, wot);

  dim3 ggrid(ROWS / 64, DMODEL / 128);
  // Q scaled by 1/sqrt(1024) = 1/32, folded into the projection epilogue.
  gemm_kernel<0><<<ggrid, 128, 0, stream>>>(xh, wqt, bq, qh, nullptr, 0.03125f);
  gemm_kernel<0><<<ggrid, 128, 0, stream>>>(xh, wkt, bk, kh, nullptr, 1.0f);
  gemm_kernel<2><<<ggrid, 128, 0, stream>>>(xh, wvt, bv, vtp, nullptr, 1.0f);

  attn_kernel<<<dim3(SEQ / 32, BATCH * HEADS), 32, 0, stream>>>(qh, kh, vtp, ctx);

  gemm_kernel<1><<<ggrid, 128, 0, stream>>>(ctx, wot, bo, nullptr, attn, 1.0f);

  ln_kernel<<<ROWS, 256, 0, stream>>>(attn, x, gamma, beta, out);
}
